// GlobalFilter_59846074302963
// MI455X (gfx1250) — matrix operand order repaired and checked by reading, not compiled, not run on hardware
//
#include <hip/hip_runtime.h>
#include <cstdint>
#include <cstddef>

// GFNet global filter == per-channel 14x14 circular convolution:
//   out[b,:,:,c] = x[b,:,:,c] (circ*) w_c,  w_c = irfft2(W_c, norm="backward")
// Per-channel circulant GEMM on v_wmma_f32_16x16x32_f16; circulant B-matrices
// (77MB, f16, padded 224x224) are rebuilt each call in d_ws and stay L2-resident.
// B slabs are staged LDS-side by the Tensor Data Mover (double-buffered,
// overlapping WMMA compute), X tiles staged f32->f16 by vector loads.
// Roofline: 616MB x/out HBM traffic floor (~26us @23.3TB/s); 30 GFLOP of f16
// WMMA is far below matrix-core limits => memory bound.

#define HH   14
#define WW   14
#define NSP  196          // H*W
#define CCH  768
#define BATCH 512
#define KP   224          // padded K (7*32)
#define NPP  224          // padded N (14*16)  -> 14 N-tiles, tile 13 all-zero
#define MT   64           // batch rows per block
#define LDX  224          // LDS row stride (f16) for X tile
#define LDB  40           // LDS row stride (f16) for B slab (conflict-free, = TDM pad)
#define SBSZ (NPP * LDB)  // one B buffer, f16 elements

typedef _Float16 f16;
typedef __attribute__((ext_vector_type(8)))  _Float16 v8h;
typedef __attribute__((ext_vector_type(16))) _Float16 v16h;
typedef __attribute__((ext_vector_type(8)))  float    v8f;
typedef __attribute__((ext_vector_type(4)))  unsigned int u32x4;
typedef __attribute__((ext_vector_type(8)))  int i32x8;
typedef __attribute__((ext_vector_type(4)))  int i32x4;

// ---------------------------------------------------------------------------
// Pass 1: build per-channel circulant matrices K_T[c][n][k] = w_c[(n-k) mod]
// (N-major, matching the WMMA B-fragment layout), f16, zero-padded to 224x224.
// ---------------------------------------------------------------------------
__global__ __launch_bounds__(256) void gf_build_kernels(const float* __restrict__ cw,
                                                        f16* __restrict__ kt)
{
    __shared__ float swt[2 * HH * (WW/2 + 1)];   // 224 floats: per-channel Wr/Wi
    __shared__ float sw[NSP];                    // real 14x14 conv kernel
    __shared__ float scs[HH], ssn[HH];

    const int c   = blockIdx.x;
    const int tid = threadIdx.x;

    for (int i = tid; i < 2 * HH * (WW/2 + 1); i += 256)
        swt[i] = cw[(size_t)(i >> 1) * (2 * CCH) + (size_t)c * 2 + (i & 1)];

    if (tid < HH) {
        float ang = 6.28318530717958647692f * (float)tid / (float)HH;
        scs[tid] = cosf(ang);
        ssn[tid] = sinf(ang);
    }
    __syncthreads();

    // w[u,v] = (1/196) * sum_h sum_{k=0..7} alpha_k (Wr cos - Wi sin),
    // theta = 2*pi*(h*u + k*v)/14, alpha = 2 for k=1..6 else 1  (exact irfft2)
    if (tid < NSP) {
        int u = tid / WW, v = tid % WW;
        float sum = 0.f;
        for (int h = 0; h < HH; ++h) {
            for (int k = 0; k <= WW/2; ++k) {
                int m = (h * u + k * v) % HH;
                float wr = swt[(h * (WW/2 + 1) + k) * 2 + 0];
                float wi = swt[(h * (WW/2 + 1) + k) * 2 + 1];
                float t  = wr * scs[m] - wi * ssn[m];
                sum += (k == 0 || k == WW/2) ? t : 2.f * t;
            }
        }
        sw[tid] = sum * (1.0f / (float)NSP);
    }
    __syncthreads();

    f16* dst = kt + (size_t)c * NPP * KP;
    for (int i = tid; i < NPP * KP; i += 256) {
        int n = i / KP, k = i % KP;
        float val = 0.f;
        if (n < NSP && k < NSP) {
            int du = (n / WW - k / WW + HH) % HH;
            int dv = (n % WW - k % WW + WW) % WW;
            val = sw[du * WW + dv];
        }
        dst[i] = (f16)val;
    }
}

// ---------------------------------------------------------------------------
// TDM descriptor issue: 2D tile 16 dwords x 224 rows from K_T slab -> LDS,
// with LDS padding 16DW + 4DW pad = 20DW rows (== LDB*2B = 80B, conflict-free).
// D# layout per CDNA5 ISA 8.3-8.6 (six-arg clang-23 builtin form).
// ---------------------------------------------------------------------------
__device__ __forceinline__ void tdm_load_b_slab(const f16* gsrc, unsigned int lds_off)
{
    unsigned long long ga = (unsigned long long)(size_t)gsrc;
    u32x4 g0;
    g0[0] = 1u;                                   // count=1, user descriptor
    g0[1] = lds_off;                              // lds_addr (bytes)
    g0[2] = (unsigned int)(ga & 0xFFFFFFFFull);   // global_addr[31:0]
    g0[3] = (unsigned int)((ga >> 32) & 0x1FFFFFFull) | (2u << 30); // addr[56:32], type=2

    i32x8 g1;
    g1[0] = (int)((2u << 16)        // data_size = 4B
                | (1u << 20)        // pad_enable
                | (3u << 22)        // pad_interval: 16 DW
                | (3u << 25));      // pad_amount:   4 DW
    g1[1] = (int)(112u << 16);      // tensor_dim0 = 112 DW (lo16 in [31:16])
    g1[2] = (int)(224u << 16);      // tensor_dim0 hi=0 | tensor_dim1 = 224 (lo16)
    g1[3] = (int)(16u << 16);       // tensor_dim1 hi=0 | tile_dim0 = 16 DW
    g1[4] = (int)224u;              // tile_dim1 = 224 rows, tile_dim2 = 0
    g1[5] = (int)112u;              // tensor_dim0_stride = 112 DW (448B rows)
    g1[6] = 0;                      // stride0 hi | tensor_dim1_stride lo
    g1[7] = 0;                      // tensor_dim1_stride hi

    i32x4 gz4 = {0, 0, 0, 0};              // groups 2/3 unused (2D tile)
    i32x8 gz8 = {0, 0, 0, 0, 0, 0, 0, 0};  // extra group (clang-23 form), unused
    __builtin_amdgcn_tensor_load_to_lds(g0, g1, gz4, gz4, gz8, 0);
}

// ---------------------------------------------------------------------------
// Pass 2: per-channel GEMM  Out_c(64x196) = X_c(64x224) * B_c(224x224-padded)
// 8 waves: 4 M-tiles x 2 N-groups of 7 tiles; 7 K-steps of 32 with WMMA f16.
// ---------------------------------------------------------------------------
__global__ __launch_bounds__(256) void gf_conv_gemm(const float* __restrict__ x,
                                                    const f16* __restrict__ kt,
                                                    float* __restrict__ out)
{
    __shared__ alignas(16) f16 sx[MT * LDX];     // 28,672 B X tile (f16, zero-pad K)
    __shared__ alignas(16) f16 sb[2 * SBSZ];     // 35,840 B B slabs (double buffer)

    const int c    = blockIdx.x;
    const int mt   = blockIdx.y;
    const int tid  = threadIdx.x;
    const int lane = tid & 31;
    const int wave = __builtin_amdgcn_readfirstlane(tid >> 5);  // SGPR-uniform
    const int wm   = wave & 3;     // M sub-tile (16 rows each)
    const int wn   = wave >> 2;    // N group: tiles [0..6] / [7..13]

    const f16* kch = kt + (size_t)c * NPP * KP;
    const unsigned int sb_base = (unsigned int)(size_t)&sb[0];

    // kick off TDM for slab 0 while we stage X
    if (wave == 0)
        tdm_load_b_slab(kch, sb_base);

    // ---- stage X tile: strided f32 gather -> f16 LDS, zero-pad K 196..223 ----
    const float* xc = x + (size_t)(mt * MT) * NSP * CCH + c;
    for (int i = tid; i < MT * LDX; i += 256) {
        int m = i / LDX, k = i % LDX;
        float v = 0.f;
        if (k < NSP) v = xc[(size_t)(m * NSP + k) * CCH];
        sx[m * LDX + k] = (f16)v;
    }

    if (wave == 0) __builtin_amdgcn_s_wait_tensorcnt(0);
    __syncthreads();

    v8f acc[7] = {};
    const int m0   = wm * 16;
    const int lrow = lane & 15;
    const int sel  = lane >> 4;
    const int nt0  = wn * 7;

    for (int kk = 0; kk < KP / 32; ++kk) {
        // prefetch next slab via TDM into the other buffer (overlaps WMMAs)
        if (wave == 0 && kk + 1 < KP / 32)
            tdm_load_b_slab(kch + (kk + 1) * 32, sb_base + ((kk + 1) & 1) * (SBSZ * 2));

        const f16* sbb = sb + (kk & 1) * SBSZ;
        const int  k0  = kk * 32;

        // A fragment (16x32 f16), ISA 7.12.2 interleave:
        // lanes 0-15: K [0..7],[16..23]; lanes 16-31: K [8..15],[24..31]
        const v8h a_lo = *(const v8h*)&sx[(m0 + lrow) * LDX + k0 + sel * 8];
        const v8h a_hi = *(const v8h*)&sx[(m0 + lrow) * LDX + k0 + 16 + sel * 8];
        v16h a = __builtin_shufflevector(a_lo, a_hi,
                                         0,1,2,3,4,5,6,7,8,9,10,11,12,13,14,15);
        #pragma unroll
        for (int j = 0; j < 7; ++j) {
            // B fragment (32x16): lanes 0-15 K [0..7],[16..23]; lanes 16-31 K [8..15],[24..31]
            const v8h b_lo = *(const v8h*)&sbb[((nt0 + j) * 16 + lrow) * LDB + sel * 8];
            const v8h b_hi = *(const v8h*)&sbb[((nt0 + j) * 16 + lrow) * LDB + 16 + sel * 8];
            v16h b = __builtin_shufflevector(b_lo, b_hi,
                                             0,1,2,3,4,5,6,7,8,9,10,11,12,13,14,15);
            acc[j] = __builtin_amdgcn_wmma_f32_16x16x32_f16(
                         false, a, false, b, (short)0, acc[j], false, false);
        }

        if (wave == 0 && kk + 1 < KP / 32) __builtin_amdgcn_s_wait_tensorcnt(0);
        __syncthreads();
    }

    // ---- write back: C/D layout (VGPR r: M = r + 8*sel; N = lane%16) ----
    float* outc = out + c;
    #pragma unroll
    for (int j = 0; j < 7; ++j) {
        int p = (nt0 + j) * 16 + lrow;       // spatial output index
        if (p < NSP) {
            #pragma unroll
            for (int r = 0; r < 8; ++r) {
                int gm = mt * MT + m0 + sel * 8 + r;   // batch row
                outc[((size_t)gm * NSP + p) * CCH] = acc[j][r];
            }
        }
    }
}

// ---------------------------------------------------------------------------
extern "C" void kernel_launch(void* const* d_in, const int* in_sizes, int n_in,
                              void* d_out, int out_size, void* d_ws, size_t ws_size,
                              hipStream_t stream)
{
    (void)in_sizes; (void)n_in; (void)out_size; (void)ws_size;
    const float* x  = (const float*)d_in[0];   // (512, 196, 768) f32
    const float* cw = (const float*)d_in[1];   // (14, 8, 768, 2) f32
    float* out = (float*)d_out;                // (512, 196, 768) f32
    f16* kt = (f16*)d_ws;  // needs 768*224*224*2 = 77,070,336 B of scratch

    gf_build_kernels<<<CCH, 256, 0, stream>>>(cw, kt);
    gf_conv_gemm<<<dim3(CCH, BATCH / MT), 256, 0, stream>>>(x, kt, out);
}
